// VectorQuantizer_17669495456259
// MI455X (gfx1250) — matrix operand order repaired and checked by reading, not compiled, not run on hardware
//
#include <hip/hip_runtime.h>
#include <hip/hip_bf16.h>
#include <math.h>

typedef __attribute__((ext_vector_type(16))) _Float16 v16h;
typedef __attribute__((ext_vector_type(8)))  float    v8f;

#define CHW   262144      // C*H*W
#define HW    4096        // H*W
#define NTOK  65536       // B*H*W tokens
#define NUMEL 4194304     // B*C*H*W
#define NBLOCKS 256       // 256 tokens per block (4 waves * 4 tiles * 16)

// workspace layout (in floats)
#define WS_INV  0         // 64   : 1/clip(std,eps) per channel
#define WS_CNT  64        // 1024 : histogram counts (float)
#define WS_PART 1088      // 256  : per-block SSE partials
#define WS_E2   1344      // 1024 : ||e_k||^2          (contiguous with cb below)
#define WS_CB   2368      // f16 codebook 1024x64 row-major (131072 bytes)

// ---------------- per-channel std (ddof=1) ----------------
__global__ __launch_bounds__(256) void vq_std_kernel(const float* __restrict__ ze,
                                                     float* __restrict__ ws) {
  const int c = blockIdx.x;          // 64 blocks, one channel each
  const int t = threadIdx.x;
  float s = 0.f, s2 = 0.f;
  const float* base = ze + (size_t)c * HW;
  for (int i = t; i < NTOK; i += 256) {
    int b = i >> 12, r = i & (HW - 1);
    float v = base[(size_t)b * CHW + r];
    s += v; s2 += v * v;
  }
  __shared__ float ss[256], sq[256];
  ss[t] = s; sq[t] = s2; __syncthreads();
  for (int o = 128; o > 0; o >>= 1) {
    if (t < o) { ss[t] += ss[t + o]; sq[t] += sq[t + o]; }
    __syncthreads();
  }
  if (t == 0) {
    float mean = ss[0] * (1.0f / (float)NTOK);
    float var  = (sq[0] - (float)NTOK * mean * mean) * (1.0f / (float)(NTOK - 1));
    float sd   = fmaxf(sqrtf(fmaxf(var, 0.f)), 1e-5f);
    ws[WS_INV + c] = 1.0f / sd;
  }
}

// ------------- f16 codebook + ||e||^2 + zero counters -------------
__global__ __launch_bounds__(256) void vq_prep_kernel(const float* __restrict__ ew,
                                                      float* __restrict__ ws) {
  const int k = blockIdx.x * 256 + threadIdx.x;   // 4 blocks -> 1024 rows
  _Float16* cb = (_Float16*)(ws + WS_CB);
  const float* row = ew + (size_t)k * 64;
  float e2 = 0.f;
  #pragma unroll
  for (int c = 0; c < 64; ++c) {
    float v = row[c];
    e2 += v * v;
    cb[(size_t)k * 64 + c] = (_Float16)v;
  }
  ws[WS_E2 + k] = e2;
  ws[WS_CNT + k] = 0.f;
}

// ---------------- fused GEMM + argmin + gather + loss ----------------
// 128 threads = 4 waves; each wave owns 4 token tiles (64 tokens); 256 blocks.
__global__ __launch_bounds__(128) void vq_main_kernel(const float* __restrict__ ze,
                                                      const float* __restrict__ ew,
                                                      float* __restrict__ ws,
                                                      float* __restrict__ out) {
  // LDS: [0,4096) = ||e||^2 ; [4096,135168) = codebook pre-permuted into WMMA
  // B-fragment order: byte = 4096 + ct*2048 + frag*1024 + lane*32
  __shared__ __align__(32) char smem[135168];
  __shared__ float s_inv[64];
  __shared__ int   s_idx[256];     // 4 waves * 64 tokens
  __shared__ float s_sse[4];

  const int tid  = threadIdx.x;
  const int lane = tid & 31;
  const int wv   = tid >> 5;

  if (tid < 64) s_inv[tid] = ws[WS_INV + tid];

  // ---- async-stage ||e||^2 + permuted codebook into LDS ----
  {
    const char* src_base = (const char*)(ws + WS_E2);   // e2 (4KB) then cb (128KB)
    for (int i = tid; i < 8448; i += 128) {             // 8448 16B chunks, 66/thread
      unsigned dst, src;
      if (i < 256) {                                    // ||e||^2: straight copy
        dst = (unsigned)i * 16u;
        src = (unsigned)i * 16u;
      } else {                                          // codebook: permute 16B blocks
        int d  = i - 256;
        int ct = d >> 7, rf = (d >> 6) & 1, l = (d >> 1) & 31, h2 = d & 1;
        int code = ct * 16 + (l & 15);
        src = 4096u + (unsigned)(code * 128 + rf * 64 + (l >> 4) * 32 + h2 * 16);
        dst = 4096u + (unsigned)d * 16u;
      }
      unsigned ldsa = (unsigned)(uintptr_t)(smem + dst);
      asm volatile("global_load_async_to_lds_b128 %0, %1, %2"
                   :: "v"(ldsa), "v"(src), "s"(src_base) : "memory");
    }
    asm volatile("s_wait_asynccnt 0x0" ::: "memory");
  }
  __syncthreads();

  const float* s_e2 = (const float*)smem;
  const char*  cbp  = smem + 4096;
  float* counts = ws + WS_CNT;

  const int m  = lane & 15;        // B column / token row within tile
  const int hf = lane >> 4;        // lane half
  const int gid  = blockIdx.x * 4 + wv;   // global wave id (0..1023)
  const int n0   = gid * 64;              // first of this wave's 64 tokens
  const int b    = n0 >> 12;              // 64 | 4096 -> same image
  const int rem0 = n0 & (HW - 1);

  // Build 4 pairs of A fragments (16x32 f16), normalized z
  v16h Alo[4], Ahi[4];
  #pragma unroll
  for (int t = 0; t < 4; ++t) {
    const size_t zb = (size_t)b * CHW + rem0 + t * 16 + m;
    #pragma unroll
    for (int j = 0; j < 16; ++j) {
      int k  = hf * 16 + j;
      int c0 = k, c1 = 32 + k;
      Alo[t][j] = (_Float16)(ze[zb + (size_t)c0 * HW] * s_inv[c0]);
      Ahi[t][j] = (_Float16)(ze[zb + (size_t)c1 * HW] * s_inv[c1]);
    }
  }

  float best[4][8]; int bidx[4][8];
  #pragma unroll
  for (int t = 0; t < 4; ++t)
    #pragma unroll
    for (int r = 0; r < 8; ++r) { best[t][r] = 3.4e38f; bidx[t][r] = 0; }

  // Sweep 64 code tiles from LDS: score = 0.5*||e||^2 - z.e  (minimize)
  #pragma unroll 1
  for (int ct = 0; ct < 64; ++ct) {
    v16h B0 = *(const v16h*)(cbp + ct * 2048 + lane * 32);          // conflict-free
    v16h B1 = *(const v16h*)(cbp + ct * 2048 + 1024 + lane * 32);
    const int   code = ct * 16 + m;
    const float he2  = 0.5f * s_e2[code];
    #pragma unroll
    for (int t = 0; t < 4; ++t) {
      v8f acc = {};
      acc = __builtin_amdgcn_wmma_f32_16x16x32_f16(false, Alo[t], false, B0, (short)0, acc, false, false);
      acc = __builtin_amdgcn_wmma_f32_16x16x32_f16(false, Ahi[t], false, B1, (short)0, acc, false, false);
      #pragma unroll
      for (int r = 0; r < 8; ++r) {
        float sc = he2 - acc[r];
        if (sc < best[t][r]) { best[t][r] = sc; bidx[t][r] = code; } // first-min kept
      }
    }
  }

  // Cross-lane argmin within each 16-lane half (C/D layout: vgpr r -> token r / r+8)
  #pragma unroll
  for (int t = 0; t < 4; ++t) {
    #pragma unroll
    for (int r = 0; r < 8; ++r) {
      float v = best[t][r]; int id = bidx[t][r];
      #pragma unroll
      for (int off = 1; off < 16; off <<= 1) {
        float ov = __shfl_xor(v, off, 32);
        int   oi = __shfl_xor(id, off, 32);
        if (ov < v || (ov == v && oi < id)) { v = ov; id = oi; }
      }
      if (m == 0) {
        int tok = t * 16 + r + hf * 8;
        s_idx[wv * 64 + tok] = id;
        out[(size_t)NUMEL + 2 + (size_t)(n0 + tok)] = (float)id;    // indices
        atomicAdd(&counts[id], 1.0f);                               // exact: <2^24
      }
    }
  }

  // Write z_q_st (straight-through) and accumulate SSE (same-wave LDS, in-order)
  float lsum = 0.f;
  #pragma unroll
  for (int t = 0; t < 4; ++t) {
    const int tok = t * 16 + m;
    const int id  = s_idx[wv * 64 + tok];
    const float* qrow = ew + (size_t)id * 64;
    const size_t ob = (size_t)b * CHW + rem0 + tok;
    #pragma unroll
    for (int j = 0; j < 32; ++j) {
      int c = hf + 2 * j;
      float zq = qrow[c];
      float z  = ze[ob + (size_t)c * HW] * s_inv[c];
      out[ob + (size_t)c * HW] = z + (zq - z);
      float d = z - zq;
      lsum += d * d;
    }
  }
  #pragma unroll
  for (int off = 1; off < 32; off <<= 1) lsum += __shfl_xor(lsum, off, 32);
  if (lane == 0) s_sse[wv] = lsum;
  __syncthreads();
  if (tid == 0) {
    float tt = s_sse[0] + s_sse[1] + s_sse[2] + s_sse[3];
    ws[WS_PART + blockIdx.x] = tt;   // deterministic partials
  }
}

// ---------------- loss + perplexity ----------------
__global__ __launch_bounds__(1024) void vq_final_kernel(const float* __restrict__ ws,
                                                        float* __restrict__ out) {
  const int t = threadIdx.x;
  __shared__ float rE[1024];
  __shared__ float rS[512];
  float p = ws[WS_CNT + t] * (1.0f / (float)NTOK);
  rE[t] = p * logf(fmaxf(p, 1e-10f));
  if (t < 512) rS[t] = (t < NBLOCKS) ? ws[WS_PART + t] : 0.f;
  __syncthreads();
  for (int o = 512; o > 0; o >>= 1) {
    if (t < o) rE[t] += rE[t + o];
    if (o <= 256 && t < o) rS[t] += rS[t + o];
    __syncthreads();
  }
  if (t == 0) {
    out[NUMEL]     = 1.25f * rS[0] * (1.0f / (float)NUMEL);  // commitment + codebook
    out[NUMEL + 1] = expf(-rE[0]);
  }
}

extern "C" void kernel_launch(void* const* d_in, const int* in_sizes, int n_in,
                              void* d_out, int out_size, void* d_ws, size_t ws_size,
                              hipStream_t stream) {
  (void)in_sizes; (void)n_in; (void)out_size; (void)ws_size;
  const float* ze = (const float*)d_in[0];
  const float* ew = (const float*)d_in[1];
  float* out = (float*)d_out;
  float* ws  = (float*)d_ws;

  vq_std_kernel <<<64,      256, 0, stream>>>(ze, ws);
  vq_prep_kernel<<<4,       256, 0, stream>>>(ew, ws);
  vq_main_kernel<<<NBLOCKS, 128, 0, stream>>>(ze, ew, ws, out);
  vq_final_kernel<<<1,     1024, 0, stream>>>(ws, out);
}
